// SpatioTemporalGCN_18133351924194
// MI455X (gfx1250) — matrix operand order repaired and checked by reading, not compiled, not run on hardware
//
#include <hip/hip_runtime.h>
#include <hip/hip_bf16.h>

// ============================================================================
// SpatioTemporalGCN for MI455X (gfx1250), wave32 + WMMA bf16.
//
// All linear layers / grouped temporal convs are lowered to bf16 GEMMs using
// v_wmma_f32_16x16x32_bf16 (fp32 accumulate).  BatchNorm (eval, running
// mean=0/var=1) is folded into weights+biases on device.  The two grouped
// K=5 convs (the FLOP-dominant ops, ~24 of ~33 GFLOP) use bf16 im2col with
// K=480 (multiple of 32).  Tiny ops (6x6 graph mix, node pooling, sigmoid
// attention epilogue, K=3 depthwise convs, residuals) run on VALU fp32.
//
// Input leaf order assumption: harness flattens setup_inputs() with JAX tree
// flatten => dict keys sorted alphabetically at every level:
//   0 adjacency, 1 edge_importance,
//   2..16  g1: bn_b,bn_g,ca_b1,ca_b2,ca_w1,ca_w2,dw_w,pw_b,pw_w,
//              res_b1,res_b2,res_bn_b,res_bn_g,res_w1,res_w2
//   17..31 g2: (same field order)
//   32 out_b1, 33 out_b2, 34 out_w1, 35 out_w2,
//   36..51 tcn: c1_b,c1_be,c1_g,c1_w,c2_b,c2_be,c2_g,c2_w,
//               p1_b,p1_be,p1_g,p1_w,p2_b,p2_be,p2_g,p2_w
//   52 x [16,4096,6,12]
//
// Workspace: ~550 MB (bump arena with liveness-based aliasing).
// ============================================================================

typedef __attribute__((ext_vector_type(16))) __bf16 v16bf;
typedef __attribute__((ext_vector_type(8)))  float  v8f;
typedef __hip_bfloat16 bf16_t;

#define EPSBN 1e-5f

union Frag {
  uint4 u[2];
  v16bf v;
};

// ---------------------------------------------------------------------------
// Generic bf16 WMMA GEMM:  C[M,N] = act( A[M,Kpad] @ W[Npad,Kpad]^T + bias )
// A row-major (lda=Kpad, multiple of 32), W row-major [Npad,Kpad] (weights
// pre-padded with zeros), bias f32 [Npad] pre-padded (nullable).
// act: 0=none, 1=relu, 2=sigmoid.
// Of (f32, nullable): convmode==0 -> Of[m*ldo+n]; convmode==1 ->
//   Of[((m>>12)*ctot + nbase + n)<<12 | (m&4095)]   (channel-major [B,C,T=4096])
// Ob (bf16, nullable): Ob[m*ldob+n] for all 16 tile columns (Npad coverage).
// One wave computes one 16x16 tile.  Tile guard is wave-uniform (EXEC all-1s
// around WMMA as the ISA requires).
// ---------------------------------------------------------------------------
__global__ __launch_bounds__(256) void k_gemm_bf16(
    const bf16_t* __restrict__ A, int lda,
    const bf16_t* __restrict__ W, int ldw,
    const float* __restrict__ bias,
    int M, int N, int K, int act,
    float* __restrict__ Of, int ldo,
    bf16_t* __restrict__ Ob, int ldob,
    int convmode, int nbase, int ctot) {
  const int lane = threadIdx.x & 31;
  const int wave = threadIdx.x >> 5;
  const int tiles_n = (N + 15) >> 4;
  const int tiles_m = M >> 4;
  const int tile = blockIdx.x * 8 + wave;
  if (tile >= tiles_m * tiles_n) return;  // wave-uniform
  const int tm = tile / tiles_n;
  const int tn = tile - tm * tiles_n;
  const int m0 = tm << 4;
  const int n0 = tn << 4;
  const int half = lane >> 4;  // 0 | 1
  const int l15 = lane & 15;

  // A fragment: lane holds row m0+l15; elems 0..7 = K kk+abase..+7,
  // elems 8..15 = K kk+16+abase..+23, abase = 8*half   (ISA 16-bit A 16x32)
  const bf16_t* Arow = A + (size_t)(m0 + l15) * lda + (half ? 8 : 0);
  // B fragment: lane holds col n0+l15; same K order as A (8*half base, +16 for elems 8..15)
  const bf16_t* Wrow = W + (size_t)(n0 + l15) * ldw + (half ? 8 : 0);

  v8f acc = {};
  for (int kk = 0; kk < K; kk += 32) {
    Frag fa, fb;
    fa.u[0] = *(const uint4*)(Arow + kk);
    fa.u[1] = *(const uint4*)(Arow + kk + 16);
    fb.u[0] = *(const uint4*)(Wrow + kk);
    fb.u[1] = *(const uint4*)(Wrow + kk + 16);
    acc = __builtin_amdgcn_wmma_f32_16x16x32_bf16(false, fa.v, false, fb.v,
                                                  (short)0, acc, false, false);
  }

  const int n = n0 + l15;
  const float bv = bias ? bias[n] : 0.0f;
#pragma unroll
  for (int e = 0; e < 8; ++e) {
    const int m = m0 + e + (half ? 8 : 0);
    float v = acc[e] + bv;
    if (act == 1) v = fmaxf(v, 0.0f);
    else if (act == 2) v = 1.0f / (1.0f + __expf(-v));
    if (Ob) Ob[(size_t)m * ldob + n] = __float2bfloat16(v);
    if (Of && n < N) {
      if (convmode) {
        const int b = m >> 12, t = m & 4095;
        Of[((size_t)(b * ctot + nbase + n) << 12) + t] = v;
      } else {
        Of[(size_t)m * ldo + n] = v;
      }
    }
  }
}

// ---------------------------------------------------------------------------
// Small / elementwise kernels
// ---------------------------------------------------------------------------
__global__ void k_zero16(unsigned short* __restrict__ p, size_t n) {
  size_t i = (size_t)blockIdx.x * blockDim.x + threadIdx.x;
  if (i < n) p[i] = 0;
}

__global__ void k_af(const float* __restrict__ adj, const float* __restrict__ ei,
                     float* __restrict__ af) {
  int i = blockIdx.x * blockDim.x + threadIdx.x;
  if (i < 36) af[i] = adj[i] * ei[i];
}

// f32 [N,K] -> bf16 [Npad,Kpad], optional BN-fold (row scale gamma/sqrt(1+eps))
__global__ void k_prep_w(const float* __restrict__ src, const float* __restrict__ gamma,
                         int N, int K, int Npad, int Kpad, bf16_t* __restrict__ dst) {
  size_t i = (size_t)blockIdx.x * blockDim.x + threadIdx.x;
  size_t tot = (size_t)Npad * Kpad;
  if (i >= tot) return;
  int n = (int)(i / Kpad), k = (int)(i - (size_t)n * Kpad);
  float v = 0.0f;
  if (n < N && k < K) {
    v = src[(size_t)n * K + k];
    if (gamma) v *= gamma[n] * rsqrtf(1.0f + EPSBN);
  }
  dst[i] = __float2bfloat16(v);
}

// bias' = bias*scale + beta, zero-padded to Npad
__global__ void k_prep_b(const float* __restrict__ bias, const float* __restrict__ gamma,
                         const float* __restrict__ beta, int N, int Npad,
                         float* __restrict__ dst) {
  int n = blockIdx.x * blockDim.x + threadIdx.x;
  if (n >= Npad) return;
  float v = 0.0f;
  if (n < N) {
    v = bias ? bias[n] : 0.0f;
    if (gamma) v *= gamma[n] * rsqrtf(1.0f + EPSBN);
    if (beta) v += beta[n];
  }
  dst[n] = v;
}

// xc[bt,n,d] = sum_m Af[n,m] * xd[bt,m,d]; writes bf16 [rows, 32] zero-padded
__global__ void k_graphmix(const float* __restrict__ af, const float* __restrict__ xd,
                           int cin, size_t rows, bf16_t* __restrict__ xc) {
  size_t i = (size_t)blockIdx.x * blockDim.x + threadIdx.x;
  if (i >= rows * 32) return;
  int d = (int)(i & 31);
  size_t row = i >> 5;
  int nn = (int)(row % 6);
  size_t bt = row / 6;
  float v = 0.0f;
  if (d < cin) {
    const float* base = xd + bt * 6 * cin + d;
#pragma unroll
    for (int mm = 0; mm < 6; ++mm) v += af[nn * 6 + mm] * base[mm * cin];
  }
  xc[i] = __float2bfloat16(v);
}

// pooled[bt,c] = mean over the 6 nodes of xp[bt,n,c]   -> bf16 [M2, co]
__global__ void k_pool(const float* __restrict__ xp, int co, size_t m2,
                       bf16_t* __restrict__ pl) {
  size_t i = (size_t)blockIdx.x * blockDim.x + threadIdx.x;
  if (i >= m2 * (size_t)co) return;
  int c = (int)(i % co);
  size_t bt = i / co;
  const float* base = xp + bt * 6 * co + c;
  float s = 0.0f;
#pragma unroll
  for (int mm = 0; mm < 6; ++mm) s += base[(size_t)mm * co];
  pl[i] = __float2bfloat16(s * (1.0f / 6.0f));
}

// out = relu( BN(xp*att) + r_bn ); oflat: bf16 [rows,co]; oz: f32 [B,384,4096]
__global__ void k_gcn_epi(const float* __restrict__ xp, const float* __restrict__ att,
                          const bf16_t* __restrict__ rbn,
                          const float* __restrict__ bng, const float* __restrict__ bnb,
                          int co, size_t rows, bf16_t* __restrict__ oflat,
                          float* __restrict__ oz) {
  size_t i = (size_t)blockIdx.x * blockDim.x + threadIdx.x;
  if (i >= rows * (size_t)co) return;
  int c = (int)(i % co);
  size_t row = i / co;
  size_t bt = row / 6;
  int nn = (int)(row % 6);
  float v = xp[i] * att[bt * co + c];
  v = v * (bng[c] * rsqrtf(1.0f + EPSBN)) + bnb[c];
  v += __bfloat162float(rbn[i]);
  v = fmaxf(v, 0.0f);
  if (oflat) oflat[i] = __float2bfloat16(v);
  if (oz) {
    int b = (int)(bt >> 12), t = (int)(bt & 4095);
    oz[((size_t)(b * 384 + nn * co + c) << 12) + t] = v;
  }
}

// im2col for grouped K=5 pad=2 conv, one group: IM[(b,t), ci*5+kt] (bf16)
// channel c = cbase+ci read from srcA (c<nchA) or srcB (c-nchA), both [B,C,T]
__global__ void k_im2col(const float* __restrict__ srcA, int nchA,
                         const float* __restrict__ srcB, int nchB, int cbase,
                         bf16_t* __restrict__ im) {
  size_t i = (size_t)blockIdx.x * blockDim.x + threadIdx.x;
  if (i >= (size_t)65536 * 480) return;
  int col = (int)(i % 480);
  size_t m = i / 480;
  int ci = col / 5, kt = col - ci * 5;
  int c = cbase + ci;
  int b = (int)(m >> 12);
  int t = (int)(m & 4095) + kt - 2;
  float v = 0.0f;
  if (t >= 0 && t < 4096) {
    v = (c < nchA) ? srcA[((size_t)(b * nchA + c) << 12) + t]
                   : srcB[((size_t)(b * nchB + (c - nchA)) << 12) + t];
  }
  im[i] = __float2bfloat16(v);
}

// depthwise K=3 pad=1 conv + bias + BN + relu, layout [B,192,4096]
__global__ void k_dwconv(const float* __restrict__ in, const float* __restrict__ w3,
                         const float* __restrict__ bias, const float* __restrict__ g,
                         const float* __restrict__ be, float* __restrict__ out) {
  size_t i = (size_t)blockIdx.x * blockDim.x + threadIdx.x;
  if (i >= (size_t)16 * 192 * 4096) return;
  int t = (int)(i & 4095);
  int c = (int)((i >> 12) % 192);
  const float* row = in + (i - t);
  float acc = 0.0f;
#pragma unroll
  for (int k = 0; k < 3; ++k) {
    int tt = t + k - 1;
    if (tt >= 0 && tt < 4096) acc += w3[c * 3 + k] * row[tt];
  }
  float s = g[c] * rsqrtf(1.0f + EPSBN);
  out[i] = fmaxf(s * (acc + bias[c]) + be[c], 0.0f);
}

// zt[bt,ch] = concat(x2p,x2c)[b,ch,t] + z[b,ch,t]  -> bf16 [B*T, 384]
__global__ void k_tcn_final(const float* __restrict__ x2p, const float* __restrict__ x2c,
                            const float* __restrict__ z, bf16_t* __restrict__ zt) {
  size_t i = (size_t)blockIdx.x * blockDim.x + threadIdx.x;
  if (i >= (size_t)65536 * 384) return;
  int ch = (int)(i % 384);
  size_t bt = i / 384;
  int b = (int)(bt >> 12), t = (int)(bt & 4095);
  float v = (ch < 192) ? x2p[((size_t)(b * 192 + ch) << 12) + t]
                       : x2c[((size_t)(b * 192 + ch - 192) << 12) + t];
  v += z[((size_t)(b * 384 + ch) << 12) + t];
  zt[i] = __float2bfloat16(v);
}

// ---------------------------------------------------------------------------
// Host orchestration
// ---------------------------------------------------------------------------
#define L1D(kern, n, ...) \
  kern<<<(unsigned)(((size_t)(n) + 255) / 256), 256, 0, stream>>>(__VA_ARGS__)

extern "C" void kernel_launch(void* const* d_in, const int* in_sizes, int n_in,
                              void* d_out, int out_size, void* d_ws, size_t ws_size,
                              hipStream_t stream) {
  (void)in_sizes; (void)n_in; (void)out_size; (void)ws_size;
  constexpr int Bb = 16, Tt = 4096;
  constexpr size_t M1 = (size_t)Bb * Tt * 6;  // 393216
  constexpr size_t M2 = (size_t)Bb * Tt;      // 65536
  constexpr int G1 = 2, G2 = 17, TCN = 36, IN_X = 52;

  auto F = [&](int i) -> const float* { return (const float*)d_in[i]; };

  // ---- workspace bump arena --------------------------------------------
  char* base = (char*)d_ws;
  size_t cur = 0;
  auto alloc = [&](size_t bytes) -> void* {
    cur = (cur + 255) & ~(size_t)255;
    void* r = base + cur;
    cur += bytes;
    return r;
  };

  float* Af = (float*)alloc(36 * 4);
  // prepped weights (bf16, [Npad,Kpad]) + biases (f32 [Npad])
  bf16_t* Wr1a = (bf16_t*)alloc(16 * 32 * 2);  float* Br1a = (float*)alloc(16 * 4);
  bf16_t* Wr2a = (bf16_t*)alloc(32 * 32 * 2);  float* Br2a = (float*)alloc(32 * 4);
  bf16_t* Wdwa = (bf16_t*)alloc(16 * 32 * 2);
  bf16_t* Wpwa = (bf16_t*)alloc(32 * 32 * 2);  float* Bpwa = (float*)alloc(32 * 4);
  bf16_t* Wc1a = (bf16_t*)alloc(16 * 32 * 2);  float* Bc1a = (float*)alloc(16 * 4);
  bf16_t* Wc2a = (bf16_t*)alloc(32 * 32 * 2);  float* Bc2a = (float*)alloc(32 * 4);
  bf16_t* Wr1b = (bf16_t*)alloc(16 * 32 * 2);  float* Br1b = (float*)alloc(16 * 4);
  bf16_t* Wr2b = (bf16_t*)alloc(64 * 32 * 2);  float* Br2b = (float*)alloc(64 * 4);
  bf16_t* Wdwb = (bf16_t*)alloc(32 * 32 * 2);
  bf16_t* Wpwb = (bf16_t*)alloc(64 * 32 * 2);  float* Bpwb = (float*)alloc(64 * 4);
  bf16_t* Wc1b = (bf16_t*)alloc(16 * 64 * 2);  float* Bc1b = (float*)alloc(16 * 4);
  bf16_t* Wc2b = (bf16_t*)alloc(64 * 32 * 2);  float* Bc2b = (float*)alloc(64 * 4);
  bf16_t* Wo1  = (bf16_t*)alloc(32 * 384 * 2); float* Bo1  = (float*)alloc(32 * 4);
  bf16_t* Wo2  = (bf16_t*)alloc(64 * 32 * 2);  float* Bo2  = (float*)alloc(64 * 4);
  bf16_t* Wp1  = (bf16_t*)alloc((size_t)192 * 480 * 2); float* Bp1 = (float*)alloc(192 * 4);
  bf16_t* Wp2  = (bf16_t*)alloc((size_t)192 * 480 * 2); float* Bp2 = (float*)alloc(192 * 4);

  // big buffers (with liveness-based aliasing)
  bf16_t* XB  = (bf16_t*)alloc(M1 * 32 * 2);         // x bf16 [M1,32]; dead after g1
  bf16_t* R1  = (bf16_t*)alloc(M1 * 32 * 2);         // contiguous after XB
  bf16_t* ZT  = XB;                                  // [M2,384] bf16 = XB+R1 (exact fit)
  bf16_t* RBN = (bf16_t*)alloc(M1 * 64 * 2);         // folded residual-BN
  float*  X2C = (float*)RBN;                         // alias: [16,192,4096] f32 (same size)
  float*  XD  = (float*)alloc(M1 * 32 * 4);          // depthwise-linear out
  float*  X2P = XD;                                  // alias (same size)
  bf16_t* XC  = (bf16_t*)alloc(M1 * 32 * 2);         // graph-mixed bf16
  float*  XP  = (float*)alloc(M1 * 64 * 4);          // pointwise out f32
  float*  X1C = XP;                                  // alias (half size)
  bf16_t* PL  = (bf16_t*)alloc(M2 * 64 * 2);         // pooled
  bf16_t* OO1 = PL;                                  // alias: out-MLP hidden
  bf16_t* Hb  = (bf16_t*)alloc(M2 * 32 * 2);         // attention hidden
  float*  ATT = (float*)alloc(M2 * 64 * 4);          // sigmoid attention
  bf16_t* O1  = (bf16_t*)alloc(M1 * 32 * 2);         // g1 output bf16
  float*  Z32 = (float*)alloc((size_t)Bb * 384 * Tt * 4);  // g2 out, z-layout f32
  bf16_t* IM  = (bf16_t*)alloc(M2 * 480 * 2);        // im2col (reused 8x)
  float*  X1P = (float*)alloc((size_t)Bb * 192 * Tt * 4);

  auto gemm = [&](const bf16_t* A, int lda, const bf16_t* W, int ldw,
                  const float* bias, size_t M, int N, int K, int act,
                  float* Of, int ldo, bf16_t* Ob, int ldob,
                  int convmode, int nbase, int ctot) {
    size_t tiles = (M >> 4) * (size_t)((N + 15) >> 4);
    unsigned blocks = (unsigned)((tiles + 7) / 8);
    k_gemm_bf16<<<blocks, 256, 0, stream>>>(A, lda, W, ldw, bias, (int)M, N, K,
                                            act, Of, ldo, Ob, ldob, convmode,
                                            nbase, ctot);
  };

  // ---- constant prep ----------------------------------------------------
  L1D(k_af, 64, F(0), F(1), Af);
  L1D(k_zero16, M1 * 32, (unsigned short*)R1, M1 * 32);  // cols 16..31 stay 0
  L1D(k_zero16, M2 * 32, (unsigned short*)Hb, M2 * 32);
  L1D(k_prep_w, M1 * 32, F(IN_X), nullptr, (int)M1, 12, (int)M1, 32, XB);

  // weight prep for one GCN block (field order per header comment)
  auto prep_block = [&](int pb, int cin, int co, int red,
                        bf16_t* Wr1, float* Br1, bf16_t* Wr2, float* Br2,
                        bf16_t* Wdw, bf16_t* Wpw, float* Bpw,
                        bf16_t* Wc1, float* Bc1, bf16_t* Wc2, float* Bc2) {
    int NpD = (cin + 15) & ~15;
    L1D(k_prep_w, 16 * 32, F(pb + 13), nullptr, red, cin, 16, 32, Wr1);
    L1D(k_prep_b, 16, F(pb + 9), nullptr, nullptr, red, 16, Br1);
    L1D(k_prep_w, co * 32, F(pb + 14), F(pb + 12), co, red, co, 32, Wr2);
    L1D(k_prep_b, co, F(pb + 10), F(pb + 12), F(pb + 11), co, co, Br2);
    L1D(k_prep_w, NpD * 32, F(pb + 6), nullptr, cin, cin, NpD, 32, Wdw);
    L1D(k_prep_w, co * 32, F(pb + 8), nullptr, co, cin, co, 32, Wpw);
    L1D(k_prep_b, co, F(pb + 7), nullptr, nullptr, co, co, Bpw);
    L1D(k_prep_w, 16 * co, F(pb + 4), nullptr, red, co, 16, co, Wc1);
    L1D(k_prep_b, 16, F(pb + 2), nullptr, nullptr, red, 16, Bc1);
    L1D(k_prep_w, co * 32, F(pb + 5), nullptr, co, red, co, 32, Wc2);
    L1D(k_prep_b, co, F(pb + 3), nullptr, nullptr, co, co, Bc2);
  };
  prep_block(G1, 12, 32, 8, Wr1a, Br1a, Wr2a, Br2a, Wdwa, Wpwa, Bpwa, Wc1a, Bc1a, Wc2a, Bc2a);
  prep_block(G2, 32, 64, 16, Wr1b, Br1b, Wr2b, Br2b, Wdwb, Wpwb, Bpwb, Wc1b, Bc1b, Wc2b, Bc2b);
  L1D(k_prep_w, (size_t)192 * 480, F(TCN + 11), F(TCN + 10), 192, 480, 192, 480, Wp1);
  L1D(k_prep_b, 192, F(TCN + 8), F(TCN + 10), F(TCN + 9), 192, 192, Bp1);
  L1D(k_prep_w, (size_t)192 * 480, F(TCN + 15), F(TCN + 14), 192, 480, 192, 480, Wp2);
  L1D(k_prep_b, 192, F(TCN + 12), F(TCN + 14), F(TCN + 13), 192, 192, Bp2);
  L1D(k_prep_w, (size_t)32 * 384, F(34), nullptr, 32, 384, 32, 384, Wo1);
  L1D(k_prep_b, 32, F(32), nullptr, nullptr, 32, 32, Bo1);
  L1D(k_prep_w, 64 * 32, F(35), nullptr, 64, 32, 64, 32, Wo2);
  L1D(k_prep_b, 64, F(33), nullptr, nullptr, 64, 64, Bo2);

  // ---- GCN block (all GEMMs are WMMA) -----------------------------------
  auto gblock = [&](int pb, int cin, int co, int red, const bf16_t* Ain,
                    const bf16_t* Wr1, const float* Br1, const bf16_t* Wr2,
                    const float* Br2, const bf16_t* Wdw, const bf16_t* Wpw,
                    const float* Bpw, const bf16_t* Wc1, const float* Bc1,
                    const bf16_t* Wc2, const float* Bc2,
                    bf16_t* oflat, float* oz) {
    gemm(Ain, 32, Wr1, 32, Br1, M1, red, 32, 0, nullptr, 0, R1, 32, 0, 0, 0);
    gemm(R1, 32, Wr2, 32, Br2, M1, co, 32, 0, nullptr, 0, RBN, co, 0, 0, 0);
    gemm(Ain, 32, Wdw, 32, nullptr, M1, cin, 32, 0, XD, cin, nullptr, 0, 0, 0, 0);
    L1D(k_graphmix, M1 * 32, Af, XD, cin, M1, XC);
    gemm(XC, 32, Wpw, 32, Bpw, M1, co, 32, 0, XP, co, nullptr, 0, 0, 0, 0);
    L1D(k_pool, M2 * co, XP, co, M2, PL);
    gemm(PL, co, Wc1, co, Bc1, M2, red, co, 1, nullptr, 0, Hb, 32, 0, 0, 0);
    gemm(Hb, 32, Wc2, 32, Bc2, M2, co, 32, 2, ATT, co, nullptr, 0, 0, 0, 0);
    L1D(k_gcn_epi, M1 * co, XP, ATT, RBN, F(pb + 1), F(pb + 0), co, M1, oflat, oz);
  };
  gblock(G1, 12, 32, 8, XB, Wr1a, Br1a, Wr2a, Br2a, Wdwa, Wpwa, Bpwa,
         Wc1a, Bc1a, Wc2a, Bc2a, O1, nullptr);
  gblock(G2, 32, 64, 16, O1, Wr1b, Br1b, Wr2b, Br2b, Wdwb, Wpwb, Bpwb,
         Wc1b, Bc1b, Wc2b, Bc2b, nullptr, Z32);

  // ---- TCN: grouped conv1 (im2col + WMMA GEMM, per group) ---------------
  for (int g = 0; g < 4; ++g) {
    L1D(k_im2col, M2 * 480, Z32, 384, Z32, 384, g * 96, IM);
    gemm(IM, 480, Wp1 + (size_t)g * 48 * 480, 480, Bp1 + g * 48, M2, 48, 480, 1,
         X1P, 0, nullptr, 0, 1, g * 48, 192);
  }
  L1D(k_dwconv, (size_t)Bb * 192 * Tt, X1P, F(TCN + 3), F(TCN + 0), F(TCN + 2),
      F(TCN + 1), X1C);
  for (int g = 0; g < 4; ++g) {
    L1D(k_im2col, M2 * 480, X1P, 192, X1C, 192, g * 96, IM);
    gemm(IM, 480, Wp2 + (size_t)g * 48 * 480, 480, Bp2 + g * 48, M2, 48, 480, 1,
         X2P, 0, nullptr, 0, 1, g * 48, 192);
  }
  L1D(k_dwconv, (size_t)Bb * 192 * Tt, X2P, F(TCN + 7), F(TCN + 4), F(TCN + 6),
      F(TCN + 5), X2C);
  L1D(k_tcn_final, M2 * 384, X2P, X2C, Z32, ZT);

  // ---- output MLP (WMMA) ------------------------------------------------
  gemm(ZT, 384, Wo1, 384, Bo1, M2, 32, 384, 1, nullptr, 0, OO1, 32, 0, 0, 0);
  gemm(OO1, 32, Wo2, 32, Bo2, M2, 64, 32, 1, (float*)d_out, 64, nullptr, 0, 0, 0, 0);
}
